// GraphTransformer_49864570307117
// MI455X (gfx1250) — compile-verified, matrix operand order checked by reading, not run on hardware
//
#include <hip/hip_runtime.h>
#include <cstdint>

// ---------------------------------------------------------------------------
// GraphTransformer for MI455X (gfx1250, wave32, WMMA).
//
// Heavy matmuls (QKV/O projections, QK^T, PV, final proj) run on
// v_wmma_f32_16x16x32_f16 (f16 in, fp32 accumulate).
// The per-edge MLP  e_h(a) = be2_h + sum_c relu(a*we1_c+be1_c)*we2_ch
// is a piecewise-linear scalar function of a: we precompute per-segment
// (alpha_h, beta_h) tables once per layer and evaluate e = alpha*a+beta.
// Adjacency/segment tiles are staged to LDS with CDNA5 async copies
// (global_load_async_to_lds_b128 / s_wait_asynccnt), overlapped with WMMA.
// ---------------------------------------------------------------------------

typedef _Float16 h16;
typedef __attribute__((ext_vector_type(16))) _Float16 v16h;
typedef __attribute__((ext_vector_type(8)))  float    v8f;

#define WMMA16(a, b, c) \
  __builtin_amdgcn_wmma_f32_16x16x32_f16(false, (a), false, (b), (short)0, (c), false, false)

// Load a 16x32 f16 operand fragment (A-layout) from row-major storage.
// CDNA5 16-bit A layout: lane L holds row rowBase+(L&15); k-halves
// {base..base+7} and {base+16..base+23} with base = kBase + (L>>4)*8.
// B-operand of X^T has the identical register image as A-operand of X,
// so Q@K^T loads K rows A-style, and P@V loads rows of V^T A-style.
static __device__ __forceinline__ v16h ldfrag(const h16* p, int rowBase, int kBase,
                                              int ld, int lane) {
  int r  = rowBase + (lane & 15);
  int k0 = kBase + ((lane >> 4) << 3);
  const h16* q = p + (size_t)r * ld + k0;
  union { v16h v; uint4 u[2]; } t;
  t.u[0] = *(const uint4*)(q);
  t.u[1] = *(const uint4*)(q + 16);
  return t.v;
}

// ---------------------------------------------------------------------------
// GEMM: D = A[MxK] * Bt[NnxK]^T + bias.  One wave computes a 16x64 tile
// (one A fragment amortized over 4 B fragments / 4 WMMAs per k-step).
// A, Bt are f16 row-major; output fp32 or f16 (dstF16). Nn % 64 == 0.
// ---------------------------------------------------------------------------
__global__ __launch_bounds__(32) void k_gemm(const h16* __restrict__ A,
                                             const h16* __restrict__ Bt,
                                             const float* __restrict__ bias,
                                             void* __restrict__ D,
                                             int M, int Nn, int K, int ldd, int dstF16) {
  int lane = threadIdx.x & 31;
  int ntn  = Nn >> 6;
  int tm   = (blockIdx.x / ntn) << 4;
  int tn   = (blockIdx.x % ntn) << 6;
  v8f acc[4] = {};
  for (int k = 0; k < K; k += 32) {
    v16h a = ldfrag(A, tm, k, K, lane);
#pragma unroll
    for (int j = 0; j < 4; ++j) {
      v16h b = ldfrag(Bt, tn + j * 16, k, K, lane);
      acc[j] = WMMA16(a, b, acc[j]);
    }
  }
  int rbase = tm + ((lane >> 4) << 3);
#pragma unroll
  for (int j = 0; j < 4; ++j) {
    int col = tn + j * 16 + (lane & 15);
    float bv = bias[col];
    if (dstF16) {
      h16* Df = (h16*)D;
#pragma unroll
      for (int i = 0; i < 8; ++i) Df[(size_t)(rbase + i) * ldd + col] = (h16)(acc[j][i] + bv);
    } else {
      float* Df = (float*)D;
#pragma unroll
      for (int i = 0; i < 8; ++i) Df[(size_t)(rbase + i) * ldd + col] = acc[j][i] + bv;
    }
  }
}

// Transpose + convert weights: Wt[n*K + k] = (f16) W[k*Nn + n]
__global__ void k_convT(const float* __restrict__ W, h16* __restrict__ Wt, int K, int Nn) {
  int i = blockIdx.x * blockDim.x + threadIdx.x;
  if (i >= Nn * K) return;
  int n = i / K, k = i % K;
  Wt[i] = (h16)W[(size_t)k * Nn + n];
}

// h = x (fp32) and hb = f16(x)
__global__ void k_init(const float* __restrict__ x, float* __restrict__ h,
                       h16* __restrict__ hb, int n) {
  int i = blockIdx.x * blockDim.x + threadIdx.x;
  if (i >= n) return;
  float v = x[i];
  h[i] = v;
  hb[i] = (h16)v;
}

// ---------------------------------------------------------------------------
// Edge-MLP piecewise-linear precompute (one block).
// Breakpoints bp_c = -be1_c/we1_c; sorted; per segment s and head h:
//   alpha[s][h] = sum_{c active} we1_c * we2_ch
//   beta [s][h] = be2_h + sum_{c active} be1_c * we2_ch
// ---------------------------------------------------------------------------
__global__ void k_edgeprep(const float* __restrict__ we1, const float* __restrict__ be1,
                           const float* __restrict__ we2, const float* __restrict__ be2,
                           float* __restrict__ bp_sorted,
                           float* __restrict__ alpha, float* __restrict__ beta) {
  __shared__ float bp[128], sw1[128], sb1[128], sorted[128];
  int t = threadIdx.x;
  if (t < 128) {
    float w = we1[t], b = be1[t];
    sw1[t] = w; sb1[t] = b;
    float v = (w != 0.f) ? (-b / w) : 3.0e38f;
    if (!(v == v)) v = 3.0e38f;
    bp[t] = v;
  }
  __syncthreads();
  if (t < 128) {                       // O(n^2) rank sort, n = 128
    float v = bp[t]; int rank = 0;
    for (int j = 0; j < 128; ++j) {
      float u = bp[j];
      rank += (u < v) || (u == v && j < t);
    }
    sorted[rank] = v;
  }
  __syncthreads();
  if (t < 128) bp_sorted[t] = sorted[t];
  for (int idx = t; idx < 129 * 8; idx += 256) {
    int s = idx >> 3, h = idx & 7;
    float lo = (s == 0)   ? -1.0e30f : sorted[s - 1];
    float hi = (s == 128) ?  1.0e30f : sorted[s];
    float ar = 0.5f * lo + 0.5f * hi;     // representative point inside segment
    float al = 0.f, be = 0.f;
    for (int c = 0; c < 128; ++c) {
      float w = sw1[c], b = sb1[c];
      if (ar * w + b > 0.f) {
        float w2 = we2[c * 8 + h];
        al += w * w2;
        be += b * w2;
      }
    }
    alpha[idx] = al;
    beta[idx]  = be + be2[h];
  }
}

// Per-(b,n,m) segment index via 7-step binary search over 128 sorted breakpoints.
__global__ void k_segidx(const float* __restrict__ adj, const float* __restrict__ bp_sorted,
                         uint8_t* __restrict__ seg, int total) {
  __shared__ float s[128];
  if (threadIdx.x < 128) s[threadIdx.x] = bp_sorted[threadIdx.x];
  __syncthreads();
  int i = blockIdx.x * blockDim.x + threadIdx.x;
  if (i >= total) return;
  float a = adj[i];
  int p = 0;
#pragma unroll
  for (int st = 64; st >= 1; st >>= 1) {
    int np = p + st;
    p = (s[np - 1] < a) ? np : p;        // p = count(sorted < a) in [0,128]
  }
  seg[i] = (uint8_t)p;
}

// V [B,N,1024] f16 -> Vt [B,1024,N] f16 (so PV B-operands load contiguously)
__global__ void k_vtrans(const h16* __restrict__ Vb, h16* __restrict__ Vt) {
  __shared__ h16 tile[32][33];
  int b = blockIdx.z;
  int c0 = blockIdx.x * 32, n0 = blockIdx.y * 32;
  int tx = threadIdx.x & 31, ty = threadIdx.x >> 5;   // 32x8
  for (int r = ty; r < 32; r += 8)
    tile[r][tx] = Vb[((size_t)b * 2048 + n0 + r) * 1024 + c0 + tx];
  __syncthreads();
  for (int r = ty; r < 32; r += 8)
    Vt[((size_t)b * 1024 + c0 + r) * 2048 + n0 + tx] = tile[tx][r];
}

// ---------------------------------------------------------------------------
// Flash attention: block = 8 waves = 8 heads, one 16-row n-tile.
// Per 32-wide m-chunk: adj/seg tile is async-copied to LDS (shared by all
// 8 heads) overlapped with the QK^T WMMAs; then S = QK^T*0.25 + (alpha*a+beta),
// online softmax, O += P V.
// ---------------------------------------------------------------------------
__global__ __launch_bounds__(256) void k_attn(const h16* __restrict__ Qb,
                                              const h16* __restrict__ Kb,
                                              const h16* __restrict__ Vt,
                                              const float* __restrict__ adj,
                                              const uint8_t* __restrict__ seg,
                                              const float* __restrict__ alpha,
                                              const float* __restrict__ beta,
                                              h16* __restrict__ Ob) {
  constexpr int N = 2048;
  __shared__ float s_al[129 * 8], s_be[129 * 8];
  __shared__ __align__(16) h16 s_p[8][16 * 32];     // per-wave private P staging
  __shared__ __align__(16) float   s_adj[16 * 32];  // adj tile, shared by all heads
  __shared__ __align__(16) uint8_t s_seg[16 * 32];  // segment-index tile
  int t = threadIdx.x, lane = t & 31, h = t >> 5;
  for (int i = t; i < 129 * 8; i += 256) { s_al[i] = alpha[i]; s_be[i] = beta[i]; }
  __syncthreads();

  int b     = blockIdx.x >> 7;          // N/16 = 128 tiles per batch
  int ntile = (blockIdx.x & 127) << 4;
  const h16* Qr  = Qb + (size_t)b * N * 1024;
  const h16* Kr  = Kb + (size_t)b * N * 1024;
  const h16* Vtb = Vt + (size_t)b * 1024 * N;
  const float*   adjb = adj + (size_t)b * N * N;
  const uint8_t* segb = seg + (size_t)b * N * N;

  v16h qf[4];
#pragma unroll
  for (int kk = 0; kk < 4; ++kk)
    qf[kk] = ldfrag(Qr, ntile, h * 128 + kk * 32, 1024, lane);

  v8f o[8] = {};
  float m_i[8], l_i[8];
#pragma unroll
  for (int i = 0; i < 8; ++i) { m_i[i] = -3.0e38f; l_i[i] = 0.f; }

  int half = lane >> 4, colid = lane & 15;

  for (int mc = 0; mc < N; mc += 32) {
    // --- async-stage adj (16x32 f32) + seg (16x32 u8) tiles into LDS ---
    if (t < 128) {  // 128 threads x 16B = 2KB adj tile
      int r = t >> 3, q = t & 7;
      const float* gsrc = adjb + (size_t)(ntile + r) * N + mc + q * 4;
      unsigned ldst = (unsigned)(uintptr_t)(const void*)&s_adj[r * 32 + q * 4];
      asm volatile("global_load_async_to_lds_b128 %0, %1, off"
                   :: "v"(ldst), "v"((unsigned long long)(uintptr_t)gsrc)
                   : "memory");
    }
    if (t < 32) {   // 32 threads x 16B = 512B seg tile
      int r = t >> 1, q = t & 1;
      const uint8_t* gsrc = segb + (size_t)(ntile + r) * N + mc + q * 16;
      unsigned ldst = (unsigned)(uintptr_t)(const void*)&s_seg[r * 32 + q * 16];
      asm volatile("global_load_async_to_lds_b128 %0, %1, off"
                   :: "v"(ldst), "v"((unsigned long long)(uintptr_t)gsrc)
                   : "memory");
    }
    // --- S = Q @ K^T (overlaps with the async copies above) ---
    v8f s0 = {}, s1 = {};
#pragma unroll
    for (int kk = 0; kk < 4; ++kk) {
      v16h kb0 = ldfrag(Kr, mc,      h * 128 + kk * 32, 1024, lane);
      v16h kb1 = ldfrag(Kr, mc + 16, h * 128 + kk * 32, 1024, lane);
      s0 = WMMA16(qf[kk], kb0, s0);
      s1 = WMMA16(qf[kk], kb1, s1);
    }
    asm volatile("s_wait_asynccnt 0" ::: "memory");
    __syncthreads();
    // --- scale + piecewise-linear edge bias (tiles from LDS) ---
#pragma unroll
    for (int i = 0; i < 8; ++i) {
      int r = (i + half * 8) * 32;
      float a0 = s_adj[r + colid],      a1 = s_adj[r + 16 + colid];
      int   g0 = s_seg[r + colid],      g1 = s_seg[r + 16 + colid];
      s0[i] = s0[i] * 0.25f + s_al[g0 * 8 + h] * a0 + s_be[g0 * 8 + h];
      s1[i] = s1[i] * 0.25f + s_al[g1 * 8 + h] * a1 + s_be[g1 * 8 + h];
    }
    // --- online softmax over m (rows live in 16-lane halves) ---
#pragma unroll
    for (int i = 0; i < 8; ++i) {
      float mx = fmaxf(s0[i], s1[i]);
#pragma unroll
      for (int msk = 1; msk <= 8; msk <<= 1) mx = fmaxf(mx, __shfl_xor(mx, msk, 32));
      float mn = fmaxf(m_i[i], mx);
      float sc = __expf(m_i[i] - mn);
      float p0 = __expf(s0[i] - mn);
      float p1 = __expf(s1[i] - mn);
      float rs = p0 + p1;
#pragma unroll
      for (int msk = 1; msk <= 8; msk <<= 1) rs += __shfl_xor(rs, msk, 32);
      l_i[i] = l_i[i] * sc + rs;
      m_i[i] = mn;
      s0[i] = p0; s1[i] = p1;
#pragma unroll
      for (int j = 0; j < 8; ++j) o[j][i] *= sc;
    }
    // --- re-layout P (C-layout -> A-layout) through private LDS ---
    h16* pp = &s_p[h][0];
#pragma unroll
    for (int i = 0; i < 8; ++i) {
      int r = i + half * 8;
      pp[r * 32 + colid]      = (h16)s0[i];
      pp[r * 32 + 16 + colid] = (h16)s1[i];
    }
    v16h pf = ldfrag(pp, 0, 0, 32, lane);
    // --- O += P @ V  (B-operand = A-style rows of V^T) ---
#pragma unroll
    for (int j = 0; j < 8; ++j) {
      v16h vf = ldfrag(Vtb, h * 128 + j * 16, mc, N, lane);
      o[j] = WMMA16(pf, vf, o[j]);
    }
    __syncthreads();   // protect adj/seg tiles before next chunk's re-stage
  }
  // --- normalize and store ---
#pragma unroll
  for (int i = 0; i < 8; ++i) {
    int n = ntile + i + half * 8;
    float inv = 1.f / l_i[i];
    h16* orow = Ob + (size_t)(b * N + n) * 1024 + h * 128;
#pragma unroll
    for (int j = 0; j < 8; ++j) orow[j * 16 + colid] = (h16)(o[j][i] * inv);
  }
}

// Residual add + LayerNorm; writes fp32 h and f16 hb for next layer.
__global__ void k_addln(const float* __restrict__ tmp, const float* __restrict__ g,
                        const float* __restrict__ bta, float* __restrict__ h,
                        h16* __restrict__ hb) {
  __shared__ float red[128];
  int row = blockIdx.x, t = threadIdx.x;
  float v = h[(size_t)row * 128 + t] + tmp[(size_t)row * 128 + t];
  red[t] = v; __syncthreads();
  for (int s = 64; s > 0; s >>= 1) { if (t < s) red[t] += red[t + s]; __syncthreads(); }
  float mu = red[0] / 128.f; __syncthreads();
  float d = v - mu;
  red[t] = d * d; __syncthreads();
  for (int s = 64; s > 0; s >>= 1) { if (t < s) red[t] += red[t + s]; __syncthreads(); }
  float var = red[0] / 128.f;
  float o = d * rsqrtf(var + 1e-5f) * g[t] + bta[t];
  h[(size_t)row * 128 + t] = o;
  hb[(size_t)row * 128 + t] = (h16)o;
}

// ---------------------------------------------------------------------------
extern "C" void kernel_launch(void* const* d_in, const int* in_sizes, int n_in,
                              void* d_out, int out_size, void* d_ws, size_t ws_size,
                              hipStream_t stream) {
  (void)in_sizes; (void)n_in; (void)out_size; (void)ws_size;
  const int B = 2, N = 2048, M = B * N;   // 4096 rows
  const float* x     = (const float*)d_in[0];
  const float* adj   = (const float*)d_in[1];
  const float* w_out = (const float*)d_in[58];
  const float* b_out = (const float*)d_in[59];

  char* wp = (char*)d_ws;
  auto alloc = [&](size_t bytes) -> char* {
    char* p = wp; wp += (bytes + 255) & ~(size_t)255; return p;
  };
  h16*   hb    = (h16*)  alloc((size_t)M * 128 * 2);
  float* h     = (float*)alloc((size_t)M * 128 * 4);
  h16*   Qb    = (h16*)  alloc((size_t)M * 1024 * 2);
  h16*   Kb    = (h16*)  alloc((size_t)M * 1024 * 2);
  h16*   Vb    = (h16*)  alloc((size_t)M * 1024 * 2);
  h16*   Vt    = (h16*)  alloc((size_t)M * 1024 * 2);
  h16*   aob   = (h16*)  alloc((size_t)M * 1024 * 2);
  float* tmp   = (float*)alloc((size_t)M * 128 * 4);
  h16*   wqT   = (h16*)  alloc(1024 * 128 * 2);
  h16*   wkT   = (h16*)  alloc(1024 * 128 * 2);
  h16*   wvT   = (h16*)  alloc(1024 * 128 * 2);
  h16*   woT   = (h16*)  alloc(1024 * 128 * 2);
  h16*   woutT = (h16*)  alloc(64 * 128 * 2);
  float* bp    = (float*)alloc(128 * 4);
  float* alpha = (float*)alloc(129 * 8 * 4);
  float* beta  = (float*)alloc(129 * 8 * 4);
  uint8_t* segix = (uint8_t*)alloc((size_t)B * N * N);

  k_init<<<(M * 128) / 256, 256, 0, stream>>>(x, h, hb, M * 128);

  for (int l = 0; l < 4; ++l) {
    const float* wq   = (const float*)d_in[2 + 14 * l + 0];
    const float* bq   = (const float*)d_in[2 + 14 * l + 1];
    const float* wk   = (const float*)d_in[2 + 14 * l + 2];
    const float* bk   = (const float*)d_in[2 + 14 * l + 3];
    const float* wv   = (const float*)d_in[2 + 14 * l + 4];
    const float* bv   = (const float*)d_in[2 + 14 * l + 5];
    const float* wo   = (const float*)d_in[2 + 14 * l + 6];
    const float* bo   = (const float*)d_in[2 + 14 * l + 7];
    const float* we1  = (const float*)d_in[2 + 14 * l + 8];
    const float* be1  = (const float*)d_in[2 + 14 * l + 9];
    const float* we2  = (const float*)d_in[2 + 14 * l + 10];
    const float* be2  = (const float*)d_in[2 + 14 * l + 11];
    const float* ln_g = (const float*)d_in[2 + 14 * l + 12];
    const float* ln_b = (const float*)d_in[2 + 14 * l + 13];

    k_convT<<<512, 256, 0, stream>>>(wq, wqT, 128, 1024);
    k_convT<<<512, 256, 0, stream>>>(wk, wkT, 128, 1024);
    k_convT<<<512, 256, 0, stream>>>(wv, wvT, 128, 1024);
    k_convT<<<512, 256, 0, stream>>>(wo, woT, 1024, 128);
    k_edgeprep<<<1, 256, 0, stream>>>(we1, be1, we2, be2, bp, alpha, beta);
    k_segidx<<<(B * N * N) / 256, 256, 0, stream>>>(adj, bp, segix, B * N * N);

    k_gemm<<<(M / 16) * (1024 / 64), 32, 0, stream>>>(hb, wqT, bq, Qb, M, 1024, 128, 1024, 1);
    k_gemm<<<(M / 16) * (1024 / 64), 32, 0, stream>>>(hb, wkT, bk, Kb, M, 1024, 128, 1024, 1);
    k_gemm<<<(M / 16) * (1024 / 64), 32, 0, stream>>>(hb, wvT, bv, Vb, M, 1024, 128, 1024, 1);
    k_vtrans<<<dim3(32, 64, 2), 256, 0, stream>>>(Vb, Vt);

    k_attn<<<B * (N / 16), 256, 0, stream>>>(Qb, Kb, Vt, adj, segix, alpha, beta, aob);

    k_gemm<<<(M / 16) * (128 / 64), 32, 0, stream>>>(aob, woT, bo, tmp, M, 128, 1024, 128, 0);
    k_addln<<<M, 128, 0, stream>>>(tmp, ln_g, ln_b, h, hb);
  }

  k_convT<<<(64 * 128 + 255) / 256, 256, 0, stream>>>(w_out, woutT, 128, 64);
  k_gemm<<<(M / 16) * (64 / 64), 32, 0, stream>>>(hb, woutT, b_out, d_out, M, 64, 128, 64, 0);
}
